// DeformableInception_28363964023451
// MI455X (gfx1250) — matrix operand order repaired and checked by reading, not compiled, not run on hardware
//
#include <hip/hip_runtime.h>
#include <hip/hip_bf16.h>

typedef __attribute__((ext_vector_type(16))) __bf16 v16bf;
typedef __attribute__((ext_vector_type(8)))  float  v8f;
typedef __attribute__((ext_vector_type(4)))  int    i32x4;

#define CIN   128
#define COUT  84
#define MPAD  96
#define HW    4096
#define WDIM  64
#define KK    9
#define KTOT  2304   // 2 branches * 128 ch * 9 taps
#define KBR   1152   // per-branch K
#define KC    64     // K elements staged per LDS stage
#define NSTG  (KTOT / KC)          // 36 stages
#define NBLK  128                  // pixels per workgroup (8 waves * 16)
#define PITCH (KC + 8)             // LDS row pitch (144B): conflict-free b128 frag reads
#define ACH   (MPAD * (KC*2/16))   // 96*8  = 768 16B-chunks for A panel
#define BCH   (NBLK * (KC*2/16))   // 128*8 = 1024 chunks for B panel
#define CHUNKS_PER_THREAD ((ACH + BCH) / 256)   // 7

#if defined(__has_builtin)
#if __has_builtin(__builtin_amdgcn_global_load_async_to_lds_b128)
#define HAVE_ASYNC 1
#endif
#endif

#if defined(HAVE_ASYNC)
// Builtin signature (from hipcc diagnostic): param0 = AS(1) v4i*, 4 args total.
typedef __attribute__((address_space(1))) i32x4 global_i32x4;
typedef __attribute__((address_space(3))) i32x4 lds_i32x4;
#endif

__device__ __forceinline__ void wait_async_7() {
#if defined(HAVE_ASYNC)
#if __has_builtin(__builtin_amdgcn_s_wait_asynccnt)
    __builtin_amdgcn_s_wait_asynccnt(7);
#else
    asm volatile("s_wait_asynccnt 0x7" ::: "memory");
#endif
#endif
}
__device__ __forceinline__ void wait_async_0() {
#if defined(HAVE_ASYNC)
#if __has_builtin(__builtin_amdgcn_s_wait_asynccnt)
    __builtin_amdgcn_s_wait_asynccnt(0);
#else
    asm volatile("s_wait_asynccnt 0x0" ::: "memory");
#endif
#endif
}

struct bf32B { uint4 lo, hi; };

__device__ __forceinline__ v16bf load_frag2(const void* p0, const void* p1) {
    bf32B t;
    t.lo = *reinterpret_cast<const uint4*>(p0);
    t.hi = *reinterpret_cast<const uint4*>(p1);
    return __builtin_bit_cast(v16bf, t);
}

// ---------------------------------------------------------------------------
// Kernel 1: fold 1x1 fuse conv into deformable weights.
// Weff[o][j] = sum_m wf[o, branch*84+m] * w_branch[m, jj],  rows 84..95 = 0
// j = branch*1152 + c*9 + kk  (matches w0/w1 natural [Cout][Cin][3][3] layout)
// ---------------------------------------------------------------------------
__global__ void prep_weights_kernel(const float* __restrict__ w0,
                                    const float* __restrict__ w1,
                                    const float* __restrict__ wf,
                                    __hip_bfloat16* __restrict__ Weff) {
    int idx = blockIdx.x * blockDim.x + threadIdx.x;
    if (idx >= MPAD * KTOT) return;
    int o = idx / KTOT, j = idx % KTOT;
    float acc = 0.f;
    if (o < COUT) {
        int branch = j / KBR, jj = j % KBR;
        const float* wb  = branch ? w1 : w0;
        const float* wfr = wf + o * (2 * COUT) + branch * COUT;
        #pragma unroll 4
        for (int m = 0; m < COUT; ++m)
            acc += wfr[m] * wb[m * KBR + jj];
    }
    Weff[idx] = __float2bfloat16(acc);
}

// ---------------------------------------------------------------------------
// Kernel 2: bilinear gather -> bf16 im2col for one batch (L2-resident x).
// block = 256 threads = 2 half-blocks; each half-block = one pixel,
// thread-within-half = input channel c (0..127). Taps staged in LDS.
// grid = (HW/2, 2 branches)
// ---------------------------------------------------------------------------
__global__ void gather_kernel(const float* __restrict__ x,
                              const float* __restrict__ dm0,
                              const float* __restrict__ dm1,
                              __hip_bfloat16* __restrict__ samp,
                              int b) {
    __shared__ int   s_y0[2][KK], s_x0[2][KK];
    __shared__ float s_fy[2][KK], s_fx[2][KK];

    const int t      = threadIdx.x;
    const int half   = t >> 7;
    const int c      = t & 127;
    const int branch = blockIdx.y;
    const int pix    = blockIdx.x * 2 + half;
    const int h      = pix >> 6;
    const int w      = pix & (WDIM - 1);

    if (c < KK) {
        const int kk = c;
        const float* dm = (branch ? dm1 : dm0) + (size_t)b * (2 * KK) * HW;
        float offy = dm[(kk * 2 + 0) * HW + pix];
        float offx = dm[(kk * 2 + 1) * HW + pix];
        float py = offy + (float)h + (float)(kk / 3 - 1);
        float px = offx + (float)w + (float)(kk % 3 - 1);
        float fy0 = floorf(py), fx0 = floorf(px);
        s_y0[half][kk] = (int)fy0;
        s_x0[half][kk] = (int)fx0;
        s_fy[half][kk] = py - fy0;
        s_fx[half][kk] = px - fx0;
    }
    __syncthreads();

    const float* xb = x + ((size_t)b * CIN + c) * HW;
    __hip_bfloat16* srow = samp + (size_t)pix * KTOT + branch * KBR + c * KK;

    #pragma unroll
    for (int kk = 0; kk < KK; ++kk) {
        int   y0 = s_y0[half][kk], x0 = s_x0[half][kk];
        float fy = s_fy[half][kk], fx = s_fx[half][kk];
        float v00 = 0.f, v01 = 0.f, v10 = 0.f, v11 = 0.f;
        bool yi0 = (unsigned)y0       < (unsigned)WDIM;
        bool yi1 = (unsigned)(y0 + 1) < (unsigned)WDIM;
        bool xi0 = (unsigned)x0       < (unsigned)WDIM;
        bool xi1 = (unsigned)(x0 + 1) < (unsigned)WDIM;
        if (yi0 && xi0) v00 = xb[y0 * WDIM + x0];
        if (yi0 && xi1) v01 = xb[y0 * WDIM + x0 + 1];
        if (yi1 && xi0) v10 = xb[(y0 + 1) * WDIM + x0];
        if (yi1 && xi1) v11 = xb[(y0 + 1) * WDIM + x0 + 1];
        float v = (1.f - fy) * ((1.f - fx) * v00 + fx * v01)
                +        fy  * ((1.f - fx) * v10 + fx * v11);
        srow[kk] = __float2bfloat16(v);
    }
}

// ---------------------------------------------------------------------------
// Kernel 3: LDS-tiled GEMM for one batch, V_WMMA_F32_16X16X32_BF16.
// Workgroup = 8 waves = 96x128 output macro-tile (6 M-tiles, wave = N-tile).
// Per stage: async-stage 96x64 A panel + 128x64 B panel into double-buffered
// LDS (28KB/stage), then 12 WMMAs per wave from LDS (conflict-free pitch).
// grid = (HW/NBLK = 32)
// ---------------------------------------------------------------------------
__global__ void gemm_kernel(const __hip_bfloat16* __restrict__ Weff,
                            const __hip_bfloat16* __restrict__ samp,
                            const float* __restrict__ bias,
                            float* __restrict__ out,
                            int b) {
    __shared__ unsigned short sA[2][MPAD * PITCH];   // 2 * 13824 B
    __shared__ unsigned short sB[2][NBLK * PITCH];   // 2 * 18432 B

    const int tid   = threadIdx.x;
    const int lane  = tid & 31;
    const int wave  = tid >> 5;            // 0..7 -> N-tile within block
    const int lhalf = lane >> 4;
    const int l16   = lane & 15;
    const int nbase = blockIdx.x * NBLK;

    // stage copy: 1792 16B-chunks, 7 per thread (A: 768, B: 1024)
    auto stage_copy = [&](int st, int buf) {
        const int k0 = st * KC;
        #pragma unroll
        for (int i = 0; i < CHUNKS_PER_THREAD; ++i) {
            int u = tid + i * 256;
            const __hip_bfloat16* g;
            unsigned short* l;
            if (u < ACH) {
                int row = u >> 3, ch = u & 7;
                g = Weff + (size_t)row * KTOT + k0 + ch * 8;
                l = &sA[buf][row * PITCH + ch * 8];
            } else {
                int v = u - ACH, row = v >> 3, ch = v & 7;
                g = samp + (size_t)(nbase + row) * KTOT + k0 + ch * 8;
                l = &sB[buf][row * PITCH + ch * 8];
            }
#if defined(HAVE_ASYNC)
            __builtin_amdgcn_global_load_async_to_lds_b128(
                (global_i32x4*)g, (lds_i32x4*)l, 0, 0);
#else
            *reinterpret_cast<uint4*>(l) = *reinterpret_cast<const uint4*>(g);
#endif
        }
    };

    v8f acc[6];
    #pragma unroll
    for (int mt = 0; mt < 6; ++mt) acc[mt] = v8f{};

    stage_copy(0, 0);

    for (int s = 0; s < NSTG; ++s) {
        const int buf = s & 1;
        if (s + 1 < NSTG) {
            stage_copy(s + 1, buf ^ 1);
            wait_async_7();          // previous stage's 7 ops complete
        } else {
            wait_async_0();
        }
        __syncthreads();             // all waves' panel data visible

        const unsigned short* Bb = &sB[buf][(wave * 16 + l16) * PITCH + lhalf * 8];
        const unsigned short* Ab = &sA[buf][l16 * PITCH + lhalf * 8];
        #pragma unroll
        for (int kk = 0; kk < KC; kk += 32) {
            v16bf bfr = load_frag2(Bb + kk, Bb + kk + 16);
            #pragma unroll
            for (int mt = 0; mt < 6; ++mt) {
                const unsigned short* Ar = Ab + (mt * 16) * PITCH + kk;
                v16bf afr = load_frag2(Ar, Ar + 16);
                acc[mt] = __builtin_amdgcn_wmma_f32_16x16x32_bf16(
                              false, afr, false, bfr, (short)0, acc[mt],
                              false, false);
            }
        }
        __syncthreads();             // reads done before buffer reuse
    }

    // D layout: VGPR i -> M = i (lanes 0-15) or i+8 (lanes 16-31); N = lane&15
    const int n     = nbase + wave * 16 + l16;
    float* ob = out + (size_t)b * COUT * HW + n;
    #pragma unroll
    for (int mt = 0; mt < 6; ++mt) {
        const int rbase = mt * 16 + lhalf * 8;
        if (mt < 5) {
            #pragma unroll
            for (int i = 0; i < 8; ++i) {
                int row = rbase + i;
                ob[(size_t)row * HW] = acc[mt][i] + bias[row];
            }
        } else {
            #pragma unroll
            for (int i = 0; i < 8; ++i) {
                int row = rbase + i;
                if (row < COUT)
                    ob[(size_t)row * HW] = acc[mt][i] + bias[row];
            }
        }
    }
}

// ---------------------------------------------------------------------------
extern "C" void kernel_launch(void* const* d_in, const int* in_sizes, int n_in,
                              void* d_out, int out_size, void* d_ws, size_t ws_size,
                              hipStream_t stream) {
    const float* x   = (const float*)d_in[0];
    const float* dm0 = (const float*)d_in[1];
    const float* dm1 = (const float*)d_in[2];
    const float* w0  = (const float*)d_in[3];
    const float* w1  = (const float*)d_in[4];
    const float* wf  = (const float*)d_in[5];
    const float* bf  = (const float*)d_in[6];
    float* out = (float*)d_out;

    // ws layout: [0, 442368) Weff bf16[96][2304]; then samp bf16[4096][2304]
    __hip_bfloat16* Weff = (__hip_bfloat16*)d_ws;
    __hip_bfloat16* samp = (__hip_bfloat16*)((char*)d_ws + (size_t)MPAD * KTOT * 2);

    prep_weights_kernel<<<(MPAD * KTOT + 255) / 256, 256, 0, stream>>>(w0, w1, wf, Weff);

    for (int b = 0; b < 8; ++b) {
        gather_kernel<<<dim3(HW / 2, 2), 256, 0, stream>>>(x, dm0, dm1, samp, b);
        gemm_kernel<<<dim3(HW / NBLK), 256, 0, stream>>>(Weff, samp, bf, out, b);
    }
    (void)in_sizes; (void)n_in; (void)out_size; (void)ws_size;
}
